// Hypernet_52321291600667
// MI455X (gfx1250) — matrix operand order repaired and checked by reading, not compiled, not run on hardware
//
#include <hip/hip_runtime.h>
#include <hip/hip_bf16.h>

// ---------------------------------------------------------------------------
// MI455X (gfx1250) implementation of a T5 decoder layer w/ prefix-KV + adapter
// All GEMMs and attention matmuls run on v_wmma_f32_16x16x32_bf16 (wave32).
// GEMM A-tiles are staged to LDS via the gfx1250 async-to-LDS engine
// (GLOBAL_LOAD_ASYNC_TO_LDS_B64 / ASYNCcnt) with double buffering.
// ---------------------------------------------------------------------------

typedef __bf16 bf16;
typedef __attribute__((ext_vector_type(16))) bf16  v16bf;
typedef __attribute__((ext_vector_type(8)))  bf16  v8bf;
typedef __attribute__((ext_vector_type(8)))  float v8f;
typedef __attribute__((ext_vector_type(2)))  int   v2i;

constexpr int Bn = 4, Ln = 1024, Dn = 1024, Hn = 16, HDn = 64;
constexpr int Fn = 2816, An = 64, Pn = 30, NBn = 32;
constexpr int LKn  = Pn + Ln;   // 1054 keys incl. prefix
constexpr int LKPn = 1056;      // padded to multiple of 32
constexpr int MLn  = Bn * Ln;   // 4096 rows
constexpr int HHn  = Hn * HDn;  // 1024

#if defined(__gfx1250__) && __has_builtin(__builtin_amdgcn_global_load_async_to_lds_b64) && __has_builtin(__builtin_amdgcn_s_wait_asynccnt)
#define USE_ASYNC_LDS 1
typedef __attribute__((address_space(1))) v2i as1_v2i;
typedef __attribute__((address_space(3))) v2i as3_v2i;
#else
#define USE_ASYNC_LDS 0
#endif

// Copy 8 bytes global -> LDS (async DMA path when available).
__device__ __forceinline__ void stage8(const bf16* g, bf16* l) {
#if USE_ASYNC_LDS
  __builtin_amdgcn_global_load_async_to_lds_b64((as1_v2i*)(void*)g,
                                                (as3_v2i*)(void*)l, 0, 0);
#else
  *(uint2*)l = *(const uint2*)g;
#endif
}

__device__ __forceinline__ void async_wait0() {
#if USE_ASYNC_LDS
  __builtin_amdgcn_s_wait_asynccnt(0);
#endif
}

// Build a 16x32 bf16 A-fragment (contiguous k order: lane = row M (mod 16),
// lane/16 selects K-half; elems 0..7 -> K = h*16..h*16+7, elems 8..15 -> +8).
__device__ __forceinline__ v16bf ldA(const bf16* p0, const bf16* p1) {
  v8bf lo = *(const v8bf*)p0;
  v8bf hi = *(const v8bf*)p1;
  v16bf a;
#pragma unroll
  for (int i = 0; i < 8; ++i) { a[i] = lo[i]; a[i + 8] = hi[i]; }
  return a;
}

__device__ __forceinline__ v8f wmma_bf16(v16bf a, v16bf b, v8f c) {
  return __builtin_amdgcn_wmma_f32_16x16x32_bf16(false, a, false, b, (short)0, c,
                                                 false, false);
}

__device__ __forceinline__ float gelu_f(float x) {
  return 0.5f * x * (1.0f + tanhf(0.7978845608028654f * (x + 0.044715f * x * x * x)));
}

// ---------------------------------------------------------------------------
// GEMM: C[M,N] (f32) = A[M,K] (bf16, row-major) x Bt[N,K] (bf16, row-major =
// B transposed). Block = 128 threads = 4 waves; block tile = 16(M) x 256(N),
// each wave owns 16x64 (4 WMMA accumulators sharing one A fragment).
// The 16x32 A k-slice is staged once per block into LDS (async, double-
// buffered) and shared by all 4 waves. Optional batch via blockIdx.z.
// ---------------------------------------------------------------------------
__global__ void k_gemm(const bf16* __restrict__ A, const bf16* __restrict__ Bt,
                       float* __restrict__ C, int M, int N, int K,
                       long sA, long sB, long sC) {
  __shared__ __align__(16) bf16 sAt[2][16 * 32];
  const int tid = threadIdx.x;
  const int lane = tid & 31, wave = tid >> 5;
  const int half = lane >> 4, lc = lane & 15;
  const int m0 = blockIdx.y * 16;
  const int n0 = blockIdx.x * 256 + wave * 64;
  const bool active = (n0 < N);  // wave-uniform; EXEC all-ones inside
  const bf16* Ab = A + (long)blockIdx.z * sA + (long)m0 * K;
  const bf16* bp0 = Bt + (long)blockIdx.z * sB + (long)(n0 + 0 + lc) * K;
  const bf16* bp1 = Bt + (long)blockIdx.z * sB + (long)(n0 + 16 + lc) * K;
  const bf16* bp2 = Bt + (long)blockIdx.z * sB + (long)(n0 + 32 + lc) * K;
  const bf16* bp3 = Bt + (long)blockIdx.z * sB + (long)(n0 + 48 + lc) * K;
  float* c = C + (long)blockIdx.z * sC;

  const int row = tid >> 3, seg = tid & 7;  // 16 rows x 8 x 8B = 1KB tile
  v8f a0 = {}, a1 = {}, a2 = {}, a3 = {};

  stage8(Ab + (long)row * K + seg * 4, &sAt[0][row * 32 + seg * 4]);
  int buf = 0;
  for (int kb = 0; kb < K; kb += 32) {
    async_wait0();
    __syncthreads();
    if (kb + 32 < K)  // stage next k-slice while computing this one
      stage8(Ab + (long)row * K + (kb + 32) + seg * 4,
             &sAt[buf ^ 1][row * 32 + seg * 4]);
    if (active) {
      const bf16* ar = &sAt[buf][lc * 32 + half * 16];
      v16bf af = ldA(ar, ar + 8);
      __builtin_prefetch(bp0 + kb + 128, 0, 1);  // global_prefetch_b8
      __builtin_prefetch(bp3 + kb + 128, 0, 1);
      v16bf b0 = *(const v16bf*)(bp0 + kb + half * 16);
      v16bf b1 = *(const v16bf*)(bp1 + kb + half * 16);
      v16bf b2 = *(const v16bf*)(bp2 + kb + half * 16);
      v16bf b3 = *(const v16bf*)(bp3 + kb + half * 16);
      a0 = wmma_bf16(af, b0, a0);
      a1 = wmma_bf16(af, b1, a1);
      a2 = wmma_bf16(af, b2, a2);
      a3 = wmma_bf16(af, b3, a3);
    }
    __syncthreads();
    buf ^= 1;
  }
  if (active) {
#pragma unroll
    for (int r = 0; r < 8; ++r) {
      const long mrow = (long)(m0 + r + 8 * half) * N;
      c[mrow + n0 + 0 + lc]  = a0[r];
      c[mrow + n0 + 16 + lc] = a1[r];
      c[mrow + n0 + 32 + lc] = a2[r];
      c[mrow + n0 + 48 + lc] = a3[r];
    }
  }
}

// ---------------------------------------------------------------------------
// Flash attention: one wave (block of 32) per (b, h, 16-query tile).
// q: [B,H,L,HD] bf16 (pre-scaled), k: [B,H,LKP,HD] bf16, vt: [B,H,HD,LKP] bf16
// o: [B,L,H*HD] bf16 (ready to be the A operand of the output projection).
// ---------------------------------------------------------------------------
template <bool CAUSAL, bool BIAS>
__global__ void k_attn(const bf16* __restrict__ q, const bf16* __restrict__ kk,
                       const bf16* __restrict__ vt, const float* __restrict__ relpos,
                       bf16* __restrict__ o) {
  const int lane = threadIdx.x & 31;
  const int half = lane >> 4, lc = lane & 15;
  const int q0 = blockIdx.x * 16;
  const int h = blockIdx.y, b = blockIdx.z;
  const long bh = (long)b * Hn + h;

  const bf16* qrow = q + (bh * Ln + q0 + lc) * (long)HDn;
  v16bf qa0 = ldA(qrow + half * 16, qrow + 8 + half * 16);        // hd 0..31
  v16bf qa1 = ldA(qrow + 32 + half * 16, qrow + 40 + half * 16);   // hd 32..63
  const bf16* kbase = kk + bh * (long)LKPn * HDn;
  const bf16* vbase = vt + bh * (long)HDn * LKPn;

  __shared__ __align__(64) float s_log[16][32];
  __shared__ __align__(64) bf16 s_prob[16][32];
  __shared__ float s_f[16];
  __shared__ float s_l[16];

  float m_i = -3.0e38f, l_i = 0.0f;   // meaningful in lanes 0..15
  v8f o0 = {}, o1 = {}, o2 = {}, o3 = {};

  const int kend = CAUSAL ? (Pn + q0 + 16) : LKn;
  const int nch = (kend + 31) >> 5;

  for (int ch = 0; ch < nch; ++ch) {
    const int kb = ch * 32;
    v8f cc0 = {}, cc1 = {};
#pragma unroll
    for (int sub = 0; sub < 2; ++sub) {
      const int key = kb + sub * 16 + lc;
      const int keyc = key < LKn ? key : (LKn - 1);
      const bf16* krow = kbase + (long)keyc * HDn;
      v16bf b0 = *(const v16bf*)(krow + half * 16);
      v16bf b1 = *(const v16bf*)(krow + 32 + half * 16);
      v8f c = {};
      c = wmma_bf16(qa0, b0, c);   // contract hd 0..31
      c = wmma_bf16(qa1, b1, c);   // contract hd 32..63
#pragma unroll
      for (int r = 0; r < 8; ++r) {
        const int qpos = q0 + r + 8 * half;
        const bool valid = (key < LKn) &&
                           (!CAUSAL || key < Pn || (key - Pn) <= qpos);
        float lg = c[r];
        if (BIAS && valid && key >= Pn) {
          const int rel = qpos - (key - Pn);  // >= 0 under causal mask
          int bucket;
          if (rel < 16) bucket = rel;
          else {
            int lb = 16 + (int)(logf((float)rel * 0.0625f) * 7.6943727f);
            bucket = lb < NBn - 1 ? lb : NBn - 1;
          }
          lg += relpos[h * NBn + bucket];
        }
        c[r] = valid ? lg : -1.0e10f;
      }
      if (sub == 0) cc0 = c; else cc1 = c;
    }
    __syncthreads();
#pragma unroll
    for (int r = 0; r < 8; ++r) {
      s_log[r + 8 * half][lc] = cc0[r];
      s_log[r + 8 * half][16 + lc] = cc1[r];
    }
    __syncthreads();
    if (lane < 16) {                 // one lane per query row: online softmax
      float mc = -3.0e38f;
#pragma unroll
      for (int j = 0; j < 32; ++j) mc = fmaxf(mc, s_log[lane][j]);
      const float mn = fmaxf(m_i, mc);
      const float f = __expf(m_i - mn);
      float s = 0.0f;
#pragma unroll
      for (int j = 0; j < 32; ++j) {
        const float p = __expf(s_log[lane][j] - mn);
        s += p;
        s_prob[lane][j] = (bf16)p;
      }
      l_i = l_i * f + s;
      m_i = mn;
      s_f[lane] = f;
      s_l[lane] = l_i;
    }
    __syncthreads();
#pragma unroll
    for (int r = 0; r < 8; ++r) {    // rescale O accumulators
      const float f = s_f[r + 8 * half];
      o0[r] *= f; o1[r] *= f; o2[r] *= f; o3[r] *= f;
    }
    const bf16* prow = &s_prob[lc][0];
    v16bf pa = ldA(prow + half * 16, prow + 8 + half * 16);  // 16x32 probs
    const bf16* vcol = vbase + kb + half * 16;
    v16bf vb0 = *(const v16bf*)(vcol + (long)(0 + lc) * LKPn);
    v16bf vb1 = *(const v16bf*)(vcol + (long)(16 + lc) * LKPn);
    v16bf vb2 = *(const v16bf*)(vcol + (long)(32 + lc) * LKPn);
    v16bf vb3 = *(const v16bf*)(vcol + (long)(48 + lc) * LKPn);
    o0 = wmma_bf16(pa, vb0, o0);
    o1 = wmma_bf16(pa, vb1, o1);
    o2 = wmma_bf16(pa, vb2, o2);
    o3 = wmma_bf16(pa, vb3, o3);
    __syncthreads();
  }
#pragma unroll
  for (int r = 0; r < 8; ++r) {
    const int m = r + 8 * half;
    const float invl = 1.0f / s_l[m];
    bf16* orow = o + ((long)(b * Ln + q0 + m)) * HHn + h * HDn + lc;
    orow[0]  = (bf16)(o0[r] * invl);
    orow[16] = (bf16)(o1[r] * invl);
    orow[32] = (bf16)(o2[r] * invl);
    orow[48] = (bf16)(o3[r] * invl);
  }
}

// ---------------------------------------------------------------------------
// Elementwise / layout kernels
// ---------------------------------------------------------------------------
__global__ void k_rmsnorm(const float* __restrict__ x, const float* __restrict__ g,
                          bf16* __restrict__ out) {
  const long row = blockIdx.x;
  const float* xr = x + row * Dn;
  float ss = 0.0f;
  for (int i = threadIdx.x; i < Dn; i += 256) { const float v = xr[i]; ss += v * v; }
#pragma unroll
  for (int o2 = 16; o2 > 0; o2 >>= 1) ss += __shfl_xor(ss, o2, 32);
  __shared__ float red[8];
  if ((threadIdx.x & 31) == 0) red[threadIdx.x >> 5] = ss;
  __syncthreads();
  float tot = 0.0f;
#pragma unroll
  for (int j = 0; j < 8; ++j) tot += red[j];
  const float inv = rsqrtf(tot * (1.0f / Dn) + 1e-6f);
  for (int i = threadIdx.x; i < Dn; i += 256)
    out[row * Dn + i] = (bf16)(xr[i] * inv * g[i]);
}

__global__ void k_cvt(const float* __restrict__ in, bf16* __restrict__ out, long n) {
  const long i = (long)blockIdx.x * 256 + threadIdx.x;
  if (i < n) out[i] = (bf16)in[i];
}

// [R,C] f32 -> [C,R] bf16 (per batch = blockIdx.y, contiguous R*C strides)
__global__ void k_cvtT(const float* __restrict__ in, bf16* __restrict__ out,
                       int R, int C) {
  const long n = (long)R * C;
  const long i = (long)blockIdx.x * 256 + threadIdx.x;
  if (i >= n) return;
  const long base = (long)blockIdx.y * n;
  const int r = (int)(i / C), c = (int)(i % C);
  out[base + (long)c * R + r] = (bf16)in[base + i];
}

// C f32 [B*L, H*HD] -> q bf16 [B,H,L,HD] * scale
__global__ void k_pack_q(const float* __restrict__ C, bf16* __restrict__ q,
                         float scale) {
  const long i = (long)blockIdx.x * 256 + threadIdx.x;
  if (i >= (long)Bn * Hn * Ln * HDn) return;
  const int e = (int)(i % HDn);
  long t = i / HDn;
  const int l = (int)(t % Ln); t /= Ln;
  const int h = (int)(t % Hn);
  const int b = (int)(t / Hn);
  q[i] = (bf16)(C[((long)(b * Ln + l)) * HHn + h * HDn + e] * scale);
}

// K: prefix (sel) + projected keys -> bf16 [B,H,LKP,HD] (pad rows zeroed)
__global__ void k_pack_k(const float* __restrict__ C, const float* __restrict__ pref,
                         bf16* __restrict__ kout, int sel) {
  const long i = (long)blockIdx.x * 256 + threadIdx.x;
  if (i >= (long)Bn * Hn * LKPn * HDn) return;
  const int e = (int)(i % HDn);
  long t0 = i / HDn;
  const int t = (int)(t0 % LKPn); t0 /= LKPn;
  const int h = (int)(t0 % Hn);
  const int b = (int)(t0 / Hn);
  float v = 0.0f;
  if (t < Pn)       v = pref[(((long)(b * 2 + sel) * Pn + t) * Hn + h) * HDn + e];
  else if (t < LKn) v = C[((long)b * Ln + (t - Pn)) * HHn + h * HDn + e];
  kout[i] = (bf16)v;
}

// V transposed: bf16 [B,H,HD,LKP] (pad cols zeroed)
__global__ void k_pack_vT(const float* __restrict__ C, const float* __restrict__ pref,
                          bf16* __restrict__ vout, int sel) {
  const long i = (long)blockIdx.x * 256 + threadIdx.x;
  if (i >= (long)Bn * Hn * HDn * LKPn) return;
  const int t = (int)(i % LKPn);
  long t0 = i / LKPn;
  const int e = (int)(t0 % HDn); t0 /= HDn;
  const int h = (int)(t0 % Hn);
  const int b = (int)(t0 / Hn);
  float v = 0.0f;
  if (t < Pn)       v = pref[(((long)(b * 2 + sel) * Pn + t) * Hn + h) * HDn + e];
  else if (t < LKn) v = C[((long)b * Ln + (t - Pn)) * HHn + h * HDn + e];
  vout[i] = (bf16)v;
}

__global__ void k_add(const float* __restrict__ a, const float* __restrict__ b,
                      float* __restrict__ out, long n) {
  const long i = (long)blockIdx.x * 256 + threadIdx.x;
  if (i < n) out[i] = a[i] + b[i];
}

__global__ void k_gelu_mul(const float* __restrict__ h0, const float* __restrict__ h1,
                           bf16* __restrict__ out, long n) {
  const long i = (long)blockIdx.x * 256 + threadIdx.x;
  if (i < n) out[i] = (bf16)(gelu_f(h0[i]) * h1[i]);
}

// ad = gelu(scr + bd)  over [B,L,A]
__global__ void k_bias_gelu(const float* __restrict__ in, const float* __restrict__ bias,
                            bf16* __restrict__ out) {
  const long i = (long)blockIdx.x * 256 + threadIdx.x;
  if (i >= (long)Bn * Ln * An) return;
  const int a = (int)(i % An);
  const int b = (int)(i / ((long)Ln * An));
  out[i] = (bf16)gelu_f(in[i] + bias[b * An + a]);
}

// out = z + az + bu + y   over [B,L,D]
__global__ void k_final(const float* __restrict__ z, const float* __restrict__ az,
                        const float* __restrict__ bu, const float* __restrict__ y,
                        float* __restrict__ out) {
  const long i = (long)blockIdx.x * 256 + threadIdx.x;
  if (i >= (long)MLn * Dn) return;
  const int d = (int)(i % Dn);
  const int b = (int)(i / ((long)Ln * Dn));
  out[i] = z[i] + az[i] + bu[b * Dn + d] + y[i];
}

// ---------------------------------------------------------------------------
// Host orchestration
// ---------------------------------------------------------------------------
static inline int cdiv(long a, long b) { return (int)((a + b - 1) / b); }

extern "C" void kernel_launch(void* const* d_in, const int* in_sizes, int n_in,
                              void* d_out, int out_size, void* d_ws, size_t ws_size,
                              hipStream_t stream) {
  (void)in_sizes; (void)n_in; (void)out_size; (void)ws_size;
  const float* inputs  = (const float*)d_in[0];
  const float* encoded = (const float*)d_in[1];
  const float* ad_wd   = (const float*)d_in[2];
  const float* ad_wu   = (const float*)d_in[3];
  const float* ad_bd   = (const float*)d_in[4];
  const float* ad_bu   = (const float*)d_in[5];
  const float* pkey    = (const float*)d_in[6];
  const float* pval    = (const float*)d_in[7];
  const float* ln1     = (const float*)d_in[8];
  const float* ln2     = (const float*)d_in[9];
  const float* ln3     = (const float*)d_in[10];
  const float* sa_wq   = (const float*)d_in[11];
  const float* sa_wk   = (const float*)d_in[12];
  const float* sa_wv   = (const float*)d_in[13];
  const float* sa_wo   = (const float*)d_in[14];
  const float* ca_wq   = (const float*)d_in[15];
  const float* ca_wk   = (const float*)d_in[16];
  const float* ca_wv   = (const float*)d_in[17];
  const float* ca_wo   = (const float*)d_in[18];
  const float* relpos  = (const float*)d_in[19];
  const float* wi0     = (const float*)d_in[20];
  const float* wi1     = (const float*)d_in[21];
  const float* mwo     = (const float*)d_in[22];
  float* out = (float*)d_out;

  char* w = (char*)d_ws;
  size_t off = 0;
  auto alloc = [&](size_t bytes) -> char* {
    char* p = w + off;
    off = (off + bytes + 255) & ~(size_t)255;
    return p;
  };
  bf16* xn     = (bf16*)alloc((size_t)MLn * Dn * 2);
  bf16* encb   = (bf16*)alloc((size_t)MLn * Dn * 2);
  bf16* qb     = (bf16*)alloc((size_t)Bn * Hn * Ln * HDn * 2);
  bf16* kb     = (bf16*)alloc((size_t)Bn * Hn * LKPn * HDn * 2);
  bf16* vtb    = (bf16*)alloc((size_t)Bn * Hn * HDn * LKPn * 2);
  bf16* ob     = (bf16*)alloc((size_t)MLn * HHn * 2);
  bf16* hgb    = (bf16*)alloc((size_t)MLn * Fn * 2);
  bf16* adb    = (bf16*)alloc((size_t)Bn * Ln * An * 2);
  float* scr0  = (float*)alloc((size_t)MLn * Fn * 4);
  float* scr1  = (float*)alloc((size_t)MLn * Fn * 4);
  float* xbuf  = (float*)alloc((size_t)MLn * Dn * 4);
  float* ybuf  = (float*)alloc((size_t)MLn * Dn * 4);
  float* zbuf  = (float*)alloc((size_t)MLn * Dn * 4);
  float* azbuf = (float*)alloc((size_t)MLn * Dn * 4);
  bf16* wT_saq = (bf16*)alloc((size_t)Dn * HHn * 2);
  bf16* wT_sak = (bf16*)alloc((size_t)Dn * HHn * 2);
  bf16* wT_sav = (bf16*)alloc((size_t)Dn * HHn * 2);
  bf16* wT_sao = (bf16*)alloc((size_t)Dn * HHn * 2);
  bf16* wT_caq = (bf16*)alloc((size_t)Dn * HHn * 2);
  bf16* wT_cak = (bf16*)alloc((size_t)Dn * HHn * 2);
  bf16* wT_cav = (bf16*)alloc((size_t)Dn * HHn * 2);
  bf16* wT_cao = (bf16*)alloc((size_t)Dn * HHn * 2);
  bf16* wT_i0  = (bf16*)alloc((size_t)Dn * Fn * 2);
  bf16* wT_i1  = (bf16*)alloc((size_t)Dn * Fn * 2);
  bf16* wT_mo  = (bf16*)alloc((size_t)Fn * Dn * 2);
  bf16* wT_add = (bf16*)alloc((size_t)Bn * An * Dn * 2);
  bf16* wT_adu = (bf16*)alloc((size_t)Bn * Dn * An * 2);

  const long nDH = (long)Dn * HHn;
  const long nDF = (long)Dn * Fn;
  // weight transpose+convert (f32 -> bf16, [K,N] -> [N,K])
  k_cvtT<<<dim3(cdiv(nDH, 256), 1), 256, 0, stream>>>(sa_wq, wT_saq, Dn, HHn);
  k_cvtT<<<dim3(cdiv(nDH, 256), 1), 256, 0, stream>>>(sa_wk, wT_sak, Dn, HHn);
  k_cvtT<<<dim3(cdiv(nDH, 256), 1), 256, 0, stream>>>(sa_wv, wT_sav, Dn, HHn);
  k_cvtT<<<dim3(cdiv(nDH, 256), 1), 256, 0, stream>>>(sa_wo, wT_sao, HHn, Dn);
  k_cvtT<<<dim3(cdiv(nDH, 256), 1), 256, 0, stream>>>(ca_wq, wT_caq, Dn, HHn);
  k_cvtT<<<dim3(cdiv(nDH, 256), 1), 256, 0, stream>>>(ca_wk, wT_cak, Dn, HHn);
  k_cvtT<<<dim3(cdiv(nDH, 256), 1), 256, 0, stream>>>(ca_wv, wT_cav, Dn, HHn);
  k_cvtT<<<dim3(cdiv(nDH, 256), 1), 256, 0, stream>>>(ca_wo, wT_cao, HHn, Dn);
  k_cvtT<<<dim3(cdiv(nDF, 256), 1), 256, 0, stream>>>(wi0, wT_i0, Dn, Fn);
  k_cvtT<<<dim3(cdiv(nDF, 256), 1), 256, 0, stream>>>(wi1, wT_i1, Dn, Fn);
  k_cvtT<<<dim3(cdiv(nDF, 256), 1), 256, 0, stream>>>(mwo, wT_mo, Fn, Dn);
  k_cvtT<<<dim3(cdiv((long)Dn * An, 256), Bn), 256, 0, stream>>>(ad_wd, wT_add, Dn, An);
  k_cvtT<<<dim3(cdiv((long)An * Dn, 256), Bn), 256, 0, stream>>>(ad_wu, wT_adu, An, Dn);
  k_cvt<<<cdiv((long)MLn * Dn, 256), 256, 0, stream>>>(encoded, encb, (long)MLn * Dn);

  const dim3 gN1024(cdiv(HHn, 256), MLn / 16, 1), gNF(cdiv(Fn, 256), MLn / 16, 1);
  const dim3 blk128(128), blk32(32), blk256(256);
  const int gKV = cdiv((long)Bn * Hn * LKPn * HDn, 256);
  const int gQ  = cdiv((long)Bn * Hn * Ln * HDn, 256);
  const dim3 gAttn(Ln / 16, Hn, Bn);

  // ---- self-attention block ----
  k_rmsnorm<<<MLn, blk256, 0, stream>>>(inputs, ln1, xn);
  k_gemm<<<gN1024, blk128, 0, stream>>>(xn, wT_saq, scr0, MLn, HHn, Dn, 0, 0, 0);
  k_pack_q<<<gQ, blk256, 0, stream>>>(scr0, qb, 0.125f);  // HD^-0.5
  k_gemm<<<gN1024, blk128, 0, stream>>>(xn, wT_sak, scr0, MLn, HHn, Dn, 0, 0, 0);
  k_pack_k<<<gKV, blk256, 0, stream>>>(scr0, pkey, kb, 0);
  k_gemm<<<gN1024, blk128, 0, stream>>>(xn, wT_sav, scr0, MLn, HHn, Dn, 0, 0, 0);
  k_pack_vT<<<gKV, blk256, 0, stream>>>(scr0, pval, vtb, 0);
  k_attn<true, true><<<gAttn, blk32, 0, stream>>>(qb, kb, vtb, relpos, ob);
  k_gemm<<<gN1024, blk128, 0, stream>>>(ob, wT_sao, scr0, MLn, Dn, HHn, 0, 0, 0);
  k_add<<<cdiv((long)MLn * Dn, 256), blk256, 0, stream>>>(scr0, inputs, xbuf, (long)MLn * Dn);

  // ---- cross-attention block ----
  k_rmsnorm<<<MLn, blk256, 0, stream>>>(xbuf, ln2, xn);
  k_gemm<<<gN1024, blk128, 0, stream>>>(xn, wT_caq, scr0, MLn, HHn, Dn, 0, 0, 0);
  k_pack_q<<<gQ, blk256, 0, stream>>>(scr0, qb, 0.125f);
  k_gemm<<<gN1024, blk128, 0, stream>>>(encb, wT_cak, scr0, MLn, HHn, Dn, 0, 0, 0);
  k_pack_k<<<gKV, blk256, 0, stream>>>(scr0, pkey, kb, 1);
  k_gemm<<<gN1024, blk128, 0, stream>>>(encb, wT_cav, scr0, MLn, HHn, Dn, 0, 0, 0);
  k_pack_vT<<<gKV, blk256, 0, stream>>>(scr0, pval, vtb, 1);
  k_attn<false, false><<<gAttn, blk32, 0, stream>>>(qb, kb, vtb, relpos, ob);
  k_gemm<<<gN1024, blk128, 0, stream>>>(ob, wT_cao, scr0, MLn, Dn, HHn, 0, 0, 0);
  k_add<<<cdiv((long)MLn * Dn, 256), blk256, 0, stream>>>(scr0, xbuf, ybuf, (long)MLn * Dn);

  // ---- gated-GELU MLP + per-example adapter ----
  k_rmsnorm<<<MLn, blk256, 0, stream>>>(ybuf, ln3, xn);
  k_gemm<<<gNF, blk128, 0, stream>>>(xn, wT_i0, scr0, MLn, Fn, Dn, 0, 0, 0);
  k_gemm<<<gNF, blk128, 0, stream>>>(xn, wT_i1, scr1, MLn, Fn, Dn, 0, 0, 0);
  k_gelu_mul<<<cdiv((long)MLn * Fn, 256), blk256, 0, stream>>>(scr0, scr1, hgb, (long)MLn * Fn);
  k_gemm<<<gN1024, blk128, 0, stream>>>(hgb, wT_mo, zbuf, MLn, Dn, Fn, 0, 0, 0);
  // adapter down: [L,D]x[D,A] per batch
  k_gemm<<<dim3(1, Ln / 16, Bn), blk128, 0, stream>>>(
      xn, wT_add, scr0, Ln, An, Dn, (long)Ln * Dn, (long)An * Dn, (long)Ln * An);
  k_bias_gelu<<<cdiv((long)Bn * Ln * An, 256), blk256, 0, stream>>>(scr0, ad_bd, adb);
  // adapter up: [L,A]x[A,D] per batch
  k_gemm<<<dim3(cdiv(Dn, 256), Ln / 16, Bn), blk128, 0, stream>>>(
      adb, wT_adu, azbuf, Ln, Dn, An, (long)Ln * An, (long)Dn * An, (long)Ln * Dn);
  k_final<<<cdiv((long)MLn * Dn, 256), blk256, 0, stream>>>(zbuf, azbuf, ad_bu, ybuf, out);
}
